// PrunedLlamaAttention_44650480009582
// MI455X (gfx1250) — matrix operand order repaired and checked by reading, not compiled, not run on hardware
//
#include <hip/hip_runtime.h>
#include <hip/hip_bf16.h>

typedef __attribute__((ext_vector_type(16))) _Float16     v16h;
typedef __attribute__((ext_vector_type(8)))  float        v8f;
typedef __attribute__((ext_vector_type(4)))  unsigned int u32x4;
typedef __attribute__((ext_vector_type(8)))  unsigned int u32x8;

constexpr int kH      = 32;
constexpr int kKVH    = 8;
constexpr int kGroups = 4;
constexpr int kT      = 4096;
constexpr int kD      = 128;
constexpr int kBudget = 512;
constexpr int kQbuf   = 64;
constexpr int kKeepFirst = 16;
constexpr int kKeepLast  = 16;
constexpr int kHD     = kH * kD;
#define ALPHA_C     0.65f
#define MIN_DECAY_C 0.7f
#define EPS_C       1e-6f
#define INV_SQRT_D  0.08838834764831845f  // 1/sqrt(128)

static __device__ __forceinline__ v8f wmma_f16(v16h a, v16h b, v8f c) {
  // D = A(16x32 f16) * B(32x16 f16) + C(16x16 f32)
  return __builtin_amdgcn_wmma_f32_16x16x32_f16(false, a, false, b, (short)0, c,
                                                false, false);
}

// ---------------------------------------------------------------------------
// Tensor Data Mover: async 2D tile load (row-strided f16) global -> LDS.
// Builds the D# descriptor per CDNA5 ISA and issues
// TENSOR_LOAD_TO_LDS with group0 (4 SGPRs) + group1 (8 SGPRs). 2D tensors
// need only two descriptor groups. Tracked by TENSORcnt.
// ---------------------------------------------------------------------------
static __device__ __forceinline__ void tdm_load_2d_f16(unsigned lds_addr,
                                                       const void* gaddr,
                                                       unsigned tile_d0,     // elements (2B)
                                                       unsigned tile_d1,     // rows
                                                       unsigned tensor_d0,
                                                       unsigned tensor_d1,
                                                       unsigned stride_d0) { // elements
  const unsigned long long ga = (unsigned long long)(uintptr_t)gaddr;
  u32x4 g0;
  g0[0] = 1u;                                   // count=1 (valid descriptor)
  g0[1] = lds_addr;                             // LDS byte address
  g0[2] = (unsigned)(ga & 0xFFFFFFFFu);         // global_addr[31:0]
  g0[3] = ((unsigned)(ga >> 32) & 0x01FFFFFFu)  // global_addr[56:32]
          | (2u << 30);                         // type=2 ("image")
  u32x8 g1;
  g1[0] = (1u << 16);                           // workgroup_mask=0, data_size=1 (2B)
  g1[1] = (tensor_d0 & 0xFFFFu) << 16;          // tensor_dim0[15:0]
  g1[2] = (tensor_d0 >> 16) | ((tensor_d1 & 0xFFFFu) << 16);
  g1[3] = (tensor_d1 >> 16) | (tile_d0 << 16);  // tile_dim0
  g1[4] = tile_d1;                              // tile_dim1 (tile_dim2=0)
  g1[5] = stride_d0;                            // tensor_dim0_stride[31:0]
  g1[6] = 0u;                                   // stride hi / dim1_stride lo (unused: 2D)
  g1[7] = 0u;
  asm volatile("tensor_load_to_lds %0, %1" :: "s"(g0), "s"(g1) : "memory");
}

// ---------------------------------------------------------------------------
// Phase 1a: kq[h][t] = max_q dot(q_raw[h, T-64+q], k_raw[h/4, t]) / sqrt(D)
// One wave per (h, 16-position tile); WMMA 16x16x32 over D=128, 4 q-row tiles.
// ---------------------------------------------------------------------------
__global__ __launch_bounds__(32) void kq_kernel(const float* __restrict__ q_raw,
                                                const float* __restrict__ k_raw,
                                                float* __restrict__ kq) {
  const int lane = threadIdx.x;
  const int h    = blockIdx.y;
  const int t0   = blockIdx.x * 16;
  const int kvh  = h / kGroups;
  const int hi   = lane >> 4;
  const int lcol = lane & 15;

  v16h b[4];
#pragma unroll
  for (int s = 0; s < 4; ++s) {
    const float* kb = k_raw + ((size_t)kvh * kT + (t0 + lcol)) * kD + 32 * s + hi * 16;
#pragma unroll
    for (int j = 0; j < 16; ++j) b[s][j] = (_Float16)kb[j];
  }

  float colmax = -3.4e38f;
  for (int mt = 0; mt < 4; ++mt) {
    const int row = kT - kQbuf + mt * 16 + lcol;
    const float* qrow = q_raw + ((size_t)h * kT + row) * kD;
    v8f c = {};
#pragma unroll
    for (int s = 0; s < 4; ++s) {
      v16h a;
      const float* qb = qrow + 32 * s + hi * 16;
#pragma unroll
      for (int j = 0; j < 16; ++j) a[j] = (_Float16)qb[j];
      c = wmma_f16(a, b[s], c);
    }
#pragma unroll
    for (int r = 0; r < 8; ++r) colmax = fmaxf(colmax, c[r]);
  }
  colmax = fmaxf(colmax, __shfl_xor(colmax, 16, 32));
  if (lane < 16) kq[(size_t)h * kT + t0 + lane] = colmax * INV_SQRT_D;
}

// ---------------------------------------------------------------------------
// Phase 1b: per-head min/max of kq over T
// ---------------------------------------------------------------------------
__global__ __launch_bounds__(256) void kq_minmax_kernel(const float* __restrict__ kq,
                                                        float* __restrict__ kq_min,
                                                        float* __restrict__ kq_max) {
  __shared__ float smin[256];
  __shared__ float smax[256];
  const int h = blockIdx.x;
  float lo = 3.4e38f, hiv = -3.4e38f;
  for (int t = threadIdx.x; t < kT; t += 256) {
    float v = kq[(size_t)h * kT + t];
    lo = fminf(lo, v);
    hiv = fmaxf(hiv, v);
  }
  smin[threadIdx.x] = lo;
  smax[threadIdx.x] = hiv;
  __syncthreads();
  for (int s = 128; s > 0; s >>= 1) {
    if (threadIdx.x < s) {
      smin[threadIdx.x] = fminf(smin[threadIdx.x], smin[threadIdx.x + s]);
      smax[threadIdx.x] = fmaxf(smax[threadIdx.x], smax[threadIdx.x + s]);
    }
    __syncthreads();
  }
  if (threadIdx.x == 0) {
    kq_min[h] = smin[0];
    kq_max[h] = smax[0];
  }
}

// ---------------------------------------------------------------------------
// Phase 1c: value norms per (kvh, t) + per-kvh max
// ---------------------------------------------------------------------------
__global__ __launch_bounds__(256) void vn_kernel(const float* __restrict__ value,
                                                 float* __restrict__ vn) {
  const int idx = blockIdx.x * 256 + threadIdx.x;  // kvh*T + t
  if (idx >= kKVH * kT) return;
  const float* vp = value + (size_t)idx * kD;
  float s = 0.f;
#pragma unroll 4
  for (int d = 0; d < kD; ++d) s += vp[d] * vp[d];
  vn[idx] = sqrtf(s);
}

__global__ __launch_bounds__(256) void vn_max_kernel(const float* __restrict__ vn,
                                                     float* __restrict__ vn_max) {
  __shared__ float smax[256];
  const int kvh = blockIdx.x;
  float hiv = -3.4e38f;
  for (int t = threadIdx.x; t < kT; t += 256)
    hiv = fmaxf(hiv, vn[(size_t)kvh * kT + t]);
  smax[threadIdx.x] = hiv;
  __syncthreads();
  for (int s = 128; s > 0; s >>= 1) {
    if (threadIdx.x < s)
      smax[threadIdx.x] = fmaxf(smax[threadIdx.x], smax[threadIdx.x + s]);
    __syncthreads();
  }
  if (threadIdx.x == 0) vn_max[kvh] = smax[0];
}

// ---------------------------------------------------------------------------
// Phase 1d: global scores (max over heads) + always-keep inf
// ---------------------------------------------------------------------------
__global__ __launch_bounds__(256) void score_kernel(const float* __restrict__ kq,
                                                    const float* __restrict__ kq_min,
                                                    const float* __restrict__ kq_max,
                                                    const float* __restrict__ vn,
                                                    const float* __restrict__ vn_max,
                                                    float* __restrict__ scores) {
  const int t = blockIdx.x * 256 + threadIdx.x;
  if (t >= kT) return;
  const float dist  = (float)(kT - 1 - t);
  const float rate  = (1.0f - MIN_DECAY_C) / (float)(kT - 1);
  const float decay = fmaxf(1.0f - rate * dist, 0.0f);
  float g = -3.4e38f;
  for (int h = 0; h < kH; ++h) {
    const int kvh = h / kGroups;
    float kqn = (kq[(size_t)h * kT + t] - kq_min[h]) / (kq_max[h] - kq_min[h] + EPS_C);
    float vnn = vn[(size_t)kvh * kT + t] / (vn_max[kvh] + EPS_C);
    g = fmaxf(g, ALPHA_C * kqn + (1.0f - ALPHA_C) * vnn * decay);
  }
  if (t < kKeepFirst || t >= kT - kKeepLast) g = __builtin_inff();
  scores[t] = g;
}

// ---------------------------------------------------------------------------
// Phase 1e: top-512 via in-LDS bitonic sort (desc score, tie = asc index),
// then sort the 512 winners ascending by index. Single workgroup (512 thr).
// ---------------------------------------------------------------------------
__global__ __launch_bounds__(512) void topk_kernel(const float* __restrict__ scores,
                                                   int* __restrict__ retained) {
  __shared__ float s[kT];
  __shared__ int   id[kT];
  for (int i = threadIdx.x; i < kT; i += 512) {
    s[i]  = scores[i];
    id[i] = i;
  }
  __syncthreads();

  for (int k = 2; k <= kT; k <<= 1) {
    for (int j = k >> 1; j > 0; j >>= 1) {
      for (int i = threadIdx.x; i < kT; i += 512) {
        const int ixj = i ^ j;
        if (ixj > i) {
          const float si = s[i], sx = s[ixj];
          const int   ii = id[i], ix = id[ixj];
          const bool before_i = (si > sx) || (si == sx && ii < ix);
          const bool doswap = ((i & k) == 0) ? (!before_i) : before_i;
          if (doswap) {
            s[i] = sx;  s[ixj] = si;
            id[i] = ix; id[ixj] = ii;
          }
        }
      }
      __syncthreads();
    }
  }

  for (int k = 2; k <= kBudget; k <<= 1) {
    for (int j = k >> 1; j > 0; j >>= 1) {
      const int i = threadIdx.x;
      const int ixj = i ^ j;
      if (i < kBudget && ixj > i && ixj < kBudget) {
        const int a = id[i], b = id[ixj];
        const bool doswap = ((i & k) == 0) ? (a > b) : (a < b);
        if (doswap) { id[i] = b; id[ixj] = a; }
      }
      __syncthreads();
    }
  }
  for (int i = threadIdx.x; i < kBudget; i += 512) retained[i] = id[i];
}

// ---------------------------------------------------------------------------
// Phase 2a: gather retained K/V rows -> f16
// ---------------------------------------------------------------------------
__global__ __launch_bounds__(256) void gather_kernel(const float* __restrict__ key,
                                                     const float* __restrict__ value,
                                                     const int* __restrict__ retained,
                                                     _Float16* __restrict__ gk16,
                                                     _Float16* __restrict__ gv16) {
  const int idx = blockIdx.x * 256 + threadIdx.x;
  if (idx >= kKVH * kBudget * kD) return;
  const int d   = idx % kD;
  const int b   = (idx / kD) % kBudget;
  const int kvh = idx / (kD * kBudget);
  const size_t src = ((size_t)kvh * kT + retained[b]) * kD + d;
  gk16[idx] = (_Float16)key[src];
  gv16[idx] = (_Float16)value[src];
}

// ---------------------------------------------------------------------------
// Phase 2b: pruned attention. One wave per (head, 16-query tile): WMMA QK^T,
// f32 softmax with causal mask on retained positions, WMMA P@Vg.
// ---------------------------------------------------------------------------
__global__ __launch_bounds__(32) void attn_kernel(const float* __restrict__ q_post,
                                                  const _Float16* __restrict__ gk16,
                                                  const _Float16* __restrict__ gv16,
                                                  const int* __restrict__ retained,
                                                  _Float16* __restrict__ attn16) {
  __shared__ float    logits[16][kBudget];  // 32 KB
  __shared__ _Float16 probs[16][kBudget];   // 16 KB
  __shared__ int      rpos[kBudget];        // 2 KB
  const int lane = threadIdx.x;
  const int h    = blockIdx.y;
  const int q0   = blockIdx.x * 16;
  const int kvh  = h / kGroups;
  const int hi   = lane >> 4;
  const int lcol = lane & 15;

  for (int i = lane; i < kBudget; i += 32) rpos[i] = retained[i];
  __syncthreads();

  v16h qa[4];
  {
    const float* qrow = q_post + ((size_t)h * kT + q0 + lcol) * kD;
#pragma unroll
    for (int s = 0; s < 4; ++s) {
      const float* qb = qrow + 32 * s + hi * 16;
#pragma unroll
      for (int j = 0; j < 16; ++j) qa[s][j] = (_Float16)qb[j];
    }
  }

  const _Float16* gkh = gk16 + (size_t)kvh * kBudget * kD;
  for (int pt = 0; pt < kBudget / 16; ++pt) {
    const int p0 = pt * 16;
    v8f c = {};
#pragma unroll
    for (int s = 0; s < 4; ++s) {
      v16h b;
      const _Float16* kb = gkh + (size_t)(p0 + lcol) * kD + 32 * s + hi * 16;
#pragma unroll
      for (int j = 0; j < 16; ++j) b[j] = kb[j];
      c = wmma_f16(qa[s], b, c);
    }
    const int n    = p0 + lcol;
    const int kpos = rpos[n];
#pragma unroll
    for (int r = 0; r < 8; ++r) {
      const int m = r + hi * 8;
      const float v = c[r] * INV_SQRT_D;
      logits[m][n] = (kpos <= q0 + m) ? v : -__builtin_inff();
    }
  }
  __syncthreads();

  {
    const int m  = lcol;
    const int c0 = hi * (kBudget / 2);
    float mx = -__builtin_inff();
    for (int ci = c0; ci < c0 + kBudget / 2; ++ci) mx = fmaxf(mx, logits[m][ci]);
    mx = fmaxf(mx, __shfl_xor(mx, 16, 32));
    float sum = 0.f;
    for (int ci = c0; ci < c0 + kBudget / 2; ++ci) sum += __expf(logits[m][ci] - mx);
    sum += __shfl_xor(sum, 16, 32);
    const float inv = 1.0f / sum;
    for (int ci = c0; ci < c0 + kBudget / 2; ++ci)
      probs[m][ci] = (_Float16)(__expf(logits[m][ci] - mx) * inv);
  }
  __syncthreads();

  const _Float16* gvh = gv16 + (size_t)kvh * kBudget * kD;
  for (int dt = 0; dt < kD / 16; ++dt) {
    v8f c = {};
    for (int s = 0; s < kBudget / 32; ++s) {
      v16h a, b;
      const _Float16* pb = &probs[lcol][32 * s + hi * 16];
#pragma unroll
      for (int j = 0; j < 16; ++j) a[j] = pb[j];
      const _Float16* vb = gvh + (size_t)(32 * s + hi * 16) * kD + dt * 16 + lcol;
#pragma unroll
      for (int j = 0; j < 16; ++j) b[j] = vb[(size_t)j * kD];
      c = wmma_f16(a, b, c);
    }
#pragma unroll
    for (int r = 0; r < 8; ++r) {
      const int m = r + hi * 8;
      attn16[(size_t)(q0 + m) * kHD + h * kD + dt * 16 + lcol] = (_Float16)c[r];
    }
  }
}

// ---------------------------------------------------------------------------
// Phase 3a: convert w_o to f16 once (halves its streaming traffic)
// ---------------------------------------------------------------------------
__global__ __launch_bounds__(256) void cvt_wo_kernel(const float* __restrict__ w_o,
                                                     _Float16* __restrict__ wo16) {
  const int idx = blockIdx.x * 256 + threadIdx.x;
  if (idx < kHD * kHD) wo16[idx] = (_Float16)w_o[idx];
}

// ---------------------------------------------------------------------------
// Phase 3b: out(4096x4096) = attn16 @ wo16^T. LDS-tiled 128x128 block,
// K-chunks of 64, double-buffered via Tensor Data Mover async tile loads.
// Wave 0 issues TDM descriptors; all waves sync on TENSORcnt + barrier.
// 8 waves/block, each wave owns 16 rows x 128 cols (8 WMMA subtiles).
// ---------------------------------------------------------------------------
__global__ __launch_bounds__(256) void out_gemm_kernel(const _Float16* __restrict__ attn16,
                                                       const _Float16* __restrict__ wo16,
                                                       float* __restrict__ out) {
  __shared__ _Float16 ldsA[2][128][64];  // 2 x 16 KB: attn16 tile [t][k]
  __shared__ _Float16 ldsB[2][128][64];  // 2 x 16 KB: wo16 tile [j][k]
  const int lane = threadIdx.x & 31;
  const int wave = threadIdx.x >> 5;
  const int hi   = lane >> 4;
  const int lcol = lane & 15;
  const int t0   = blockIdx.y * 128;
  const int j0   = blockIdx.x * 128;
  constexpr int NK = kHD / 64;  // 64 K-chunks

  if (wave == 0) {
    tdm_load_2d_f16((unsigned)(uintptr_t)&ldsA[0][0][0],
                    attn16 + (size_t)t0 * kHD, 64, 128, kHD, kT, kHD);
    tdm_load_2d_f16((unsigned)(uintptr_t)&ldsB[0][0][0],
                    wo16 + (size_t)j0 * kHD, 64, 128, kHD, kHD, kHD);
  }

  v8f acc[8] = {};
  for (int kc = 0; kc < NK; ++kc) {
    const int cur = kc & 1;
    if (wave == 0 && kc + 1 < NK) {
      const int nxt = (kc + 1) & 1;
      tdm_load_2d_f16((unsigned)(uintptr_t)&ldsA[nxt][0][0],
                      attn16 + (size_t)t0 * kHD + (kc + 1) * 64, 64, 128, kHD, kT, kHD);
      tdm_load_2d_f16((unsigned)(uintptr_t)&ldsB[nxt][0][0],
                      wo16 + (size_t)j0 * kHD + (kc + 1) * 64, 64, 128, kHD, kHD, kHD);
    }
    if (kc + 1 < NK) __builtin_amdgcn_s_wait_tensorcnt(2);  // current stage done
    else             __builtin_amdgcn_s_wait_tensorcnt(0);  // drain
    __syncthreads();

    // A fragments for this wave's 16 rows (K-chunk of 64 = 2 WMMA k-steps)
    v16h a[2];
    const int row = wave * 16 + lcol;
#pragma unroll
    for (int s = 0; s < 2; ++s)
#pragma unroll
      for (int j = 0; j < 16; ++j)
        a[s][j] = ldsA[cur][row][32 * s + j + hi * 16];

#pragma unroll
    for (int jt = 0; jt < 8; ++jt) {
#pragma unroll
      for (int s = 0; s < 2; ++s) {
        v16h b;
#pragma unroll
        for (int j = 0; j < 16; ++j)
          b[j] = ldsB[cur][jt * 16 + lcol][32 * s + j + hi * 16];
        acc[jt] = wmma_f16(a[s], b, acc[jt]);
      }
    }
    __syncthreads();  // protect LDS buffer before TDM overwrites it
  }

#pragma unroll
  for (int jt = 0; jt < 8; ++jt)
#pragma unroll
    for (int r = 0; r < 8; ++r) {
      const int m = r + hi * 8;
      out[(size_t)(t0 + wave * 16 + m) * kHD + j0 + jt * 16 + lcol] = acc[jt][r];
    }
}

// ---------------------------------------------------------------------------
extern "C" void kernel_launch(void* const* d_in, const int* in_sizes, int n_in,
                              void* d_out, int out_size, void* d_ws, size_t ws_size,
                              hipStream_t stream) {
  (void)in_sizes; (void)n_in; (void)out_size; (void)ws_size;
  const float* q_post = (const float*)d_in[0];
  const float* q_raw  = (const float*)d_in[1];
  const float* k_raw  = (const float*)d_in[2];
  const float* key    = (const float*)d_in[3];
  const float* value  = (const float*)d_in[4];
  const float* w_o    = (const float*)d_in[5];
  float* out = (float*)d_out;

  uintptr_t p = (uintptr_t)d_ws;
  auto carve = [&p](size_t bytes) -> void* {
    uintptr_t r = (p + 255) & ~(uintptr_t)255;
    p = r + bytes;
    return (void*)r;
  };
  float*    kq       = (float*)carve((size_t)kH * kT * sizeof(float));
  float*    kq_min   = (float*)carve(kH * sizeof(float));
  float*    kq_max   = (float*)carve(kH * sizeof(float));
  float*    vn       = (float*)carve((size_t)kKVH * kT * sizeof(float));
  float*    vn_max   = (float*)carve(kKVH * sizeof(float));
  float*    scores   = (float*)carve(kT * sizeof(float));
  int*      retained = (int*)carve(kBudget * sizeof(int));
  _Float16* gk16     = (_Float16*)carve((size_t)kKVH * kBudget * kD * sizeof(_Float16));
  _Float16* gv16     = (_Float16*)carve((size_t)kKVH * kBudget * kD * sizeof(_Float16));
  _Float16* attn16   = (_Float16*)carve((size_t)kT * kHD * sizeof(_Float16));
  _Float16* wo16     = (_Float16*)carve((size_t)kHD * kHD * sizeof(_Float16));

  kq_kernel<<<dim3(kT / 16, kH), 32, 0, stream>>>(q_raw, k_raw, kq);
  kq_minmax_kernel<<<kH, 256, 0, stream>>>(kq, kq_min, kq_max);
  vn_kernel<<<(kKVH * kT + 255) / 256, 256, 0, stream>>>(value, vn);
  vn_max_kernel<<<kKVH, 256, 0, stream>>>(vn, vn_max);
  score_kernel<<<kT / 256, 256, 0, stream>>>(kq, kq_min, kq_max, vn, vn_max, scores);
  topk_kernel<<<1, 512, 0, stream>>>(scores, retained);
  gather_kernel<<<(kKVH * kBudget * kD + 255) / 256, 256, 0, stream>>>(key, value,
                                                                       retained, gk16, gv16);
  attn_kernel<<<dim3(kT / 16, kH), 32, 0, stream>>>(q_post, gk16, gv16, retained, attn16);
  cvt_wo_kernel<<<(kHD * kHD + 255) / 256, 256, 0, stream>>>(w_o, wo16);
  out_gemm_kernel<<<dim3(kHD / 128, kT / 128), 256, 0, stream>>>(attn16, wo16, out);
}
